// MultiHeadedAttentionRoped_48095043781140
// MI455X (gfx1250) — compile-verified, matrix operand order checked by reading, not run on hardware
//
#include <hip/hip_runtime.h>

// ---------------------------------------------------------------------------
// MI455X (gfx1250) multi-head attention w/ RoPE, bf16 WMMA pipeline.
// B=2, T=2048, D=1024, H=16, d_h=64.  wave32 only.
// ---------------------------------------------------------------------------

typedef __bf16 v8bf  __attribute__((ext_vector_type(8)));
typedef __bf16 v16bf __attribute__((ext_vector_type(16)));
typedef float  v8f   __attribute__((ext_vector_type(8)));
typedef unsigned int u32x4 __attribute__((ext_vector_type(4)));
typedef int          i32x4 __attribute__((ext_vector_type(4)));
typedef int          i32x8 __attribute__((ext_vector_type(8)));

#define BB 2
#define TT 2048
#define DD 1024
#define HH 16
#define DH 64
#define MM (BB * TT)   // 4096 rows

#ifndef __has_builtin
#define __has_builtin(x) 0
#endif
#if __has_builtin(__builtin_amdgcn_tensor_load_to_lds) && \
    __has_builtin(__builtin_amdgcn_s_wait_tensorcnt)
#define MHA_USE_TDM 1
#else
#define MHA_USE_TDM 0
#endif

// D = A(16x32 bf16) x B(32x16 bf16) + C(16x16 f32), wave32
__device__ __forceinline__ v8f wmma_bf16(v16bf a, v16bf b, v8f c) {
  return __builtin_amdgcn_wmma_f32_16x16x32_bf16(
      /*neg_a=*/false, a, /*neg_b=*/false, b,
      /*c_mod=*/(short)0, c, /*reuse_a=*/false, /*reuse_b=*/false);
}

// Lane fragment from a row of a 32-wide K slice.  ISA 16-bit A layout:
// elems 0..7 <- K = hi*8 + j ; elems 8..15 <- K = 16 + hi*8 + j (hi=lane>>4).
__device__ __forceinline__ v16bf frag_row(const __bf16* p, int hi) {
  v8bf lo = *(const v8bf*)(p + hi * 8);
  v8bf hv = *(const v8bf*)(p + 16 + hi * 8);
  return __builtin_shufflevector(lo, hv, 0, 1, 2, 3, 4, 5, 6, 7,
                                 8, 9, 10, 11, 12, 13, 14, 15);
}

__device__ __forceinline__ float rmax16(float x) {
  #pragma unroll
  for (int off = 1; off < 16; off <<= 1) x = fmaxf(x, __shfl_xor(x, off, 32));
  return x;
}
__device__ __forceinline__ float rsum16(float x) {
  #pragma unroll
  for (int off = 1; off < 16; off <<= 1) x += __shfl_xor(x, off, 32);
  return x;
}

#if MHA_USE_TDM
// Issue a TDM 2-D tile load: rows x 64 bf16 elements, row-major, contiguous
// (tensor_dim0_stride = 64), global -> LDS.  Wave-level op; EXEC ignored.
__device__ __forceinline__ void tdm_load_rows64(const __bf16* gsrc,
                                                unsigned lds_off, int rows) {
  unsigned long long ga = (unsigned long long)(__SIZE_TYPE__)gsrc;
  u32x4 g0;
  g0[0] = 1u;                                   // count=1, user descriptor
  g0[1] = lds_off;                              // D#.lds_addr (bytes)
  g0[2] = (unsigned)ga;                         // global_addr[31:0]
  g0[3] = (unsigned)(ga >> 32) | (2u << 30);    // global_addr[56:32] | type=2
  i32x8 g1;
  g1[0] = (1 << 16);            // workgroup_mask=0, data_size=1 (2 bytes)
  g1[1] = (DH << 16);           // tensor_dim0[15:0] = 64  (bits 63:48)
  g1[2] = (TT << 16);           // tensor_dim1[15:0] = T   (bits 95:80)
  g1[3] = (DH << 16);           // tile_dim0 = 64          (bits 127:112)
  g1[4] = rows;                 // tile_dim1 = rows        (bits 143:128)
  g1[5] = DH;                   // tensor_dim0_stride = 64 (bits 191:160)
  g1[6] = 0;
  g1[7] = 0;
  i32x4 z4 = {};
#if __clang_major__ >= 23
  i32x8 z8 = {};
  __builtin_amdgcn_tensor_load_to_lds(g0, g1, z4, z4, z8, 0);
#else
  __builtin_amdgcn_tensor_load_to_lds(g0, g1, z4, z4, 0);
#endif
}
#endif

// ---------------------------------------------------------------------------
// fp32 -> bf16 elementwise
// ---------------------------------------------------------------------------
__global__ void mha_cvt_bf16(const float* __restrict__ in,
                             __bf16* __restrict__ out, int n) {
  int i = blockIdx.x * 256 + threadIdx.x;
  if (i < n) out[i] = (__bf16)in[i];
}

// ---------------------------------------------------------------------------
// C[M,N] f32 = A[M,K] bf16 (row-major) x W[N,K] bf16 (row-major, acts as B^T)
// block = 128 threads (4 waves); block tile 128(M) x 64(N);
// per-wave tile 32(M) x 64(N): 2 A-frags x 4 B-frags = 8 WMMA per K-step,
// each B fragment reused twice.  K staged through LDS in 32-wide slices.
// ---------------------------------------------------------------------------
__global__ __launch_bounds__(128)
void mha_gemm_bf16(const __bf16* __restrict__ A, const __bf16* __restrict__ W,
                   float* __restrict__ C, int M, int N, int K) {
  __shared__ __align__(16) __bf16 sA[128 * 32];   // 8 KiB
  __shared__ __align__(16) __bf16 sB[64 * 32];    // 4 KiB

  const int m0 = blockIdx.y * 128;
  const int n0 = blockIdx.x * 64;
  const int w    = threadIdx.x >> 5;
  const int lane = threadIdx.x & 31;
  const int r  = lane & 15;
  const int hi = lane >> 4;

  v8f acc[2][4] = {};

  for (int k0 = 0; k0 < K; k0 += 32) {
    // cooperative stage in 8-elem (16B) chunks
    #pragma unroll
    for (int it = 0; it < 4; ++it) {               // A: 512 chunks
      int c = threadIdx.x + it * 128;
      int row = c >> 2, kc = c & 3;
      *(v8bf*)(sA + c * 8) =
          *(const v8bf*)(A + (size_t)(m0 + row) * K + k0 + kc * 8);
    }
    #pragma unroll
    for (int it = 0; it < 2; ++it) {               // B: 256 chunks
      int c = threadIdx.x + it * 128;
      int row = c >> 2, kc = c & 3;
      *(v8bf*)(sB + c * 8) =
          *(const v8bf*)(W + (size_t)(n0 + row) * K + k0 + kc * 8);
    }
    // prefetch next K slice into cache (global_prefetch_b8)
    if (k0 + 32 < K) {
      __builtin_prefetch(A + (size_t)(m0 + threadIdx.x) * K + k0 + 32, 0, 1);
      __builtin_prefetch(W + (size_t)(n0 + (threadIdx.x >> 1)) * K + k0 + 32,
                         0, 1);
    }
    __syncthreads();

    v16bf a0 = frag_row(sA + (w * 32 + r) * 32, hi);
    v16bf a1 = frag_row(sA + (w * 32 + 16 + r) * 32, hi);
    #pragma unroll
    for (int nt = 0; nt < 4; ++nt) {
      v16bf bf = frag_row(sB + (nt * 16 + r) * 32, hi);
      acc[0][nt] = wmma_bf16(a0, bf, acc[0][nt]);
      acc[1][nt] = wmma_bf16(a1, bf, acc[1][nt]);
    }
    __syncthreads();
  }

  // C layout: VGPR p, lane -> (m = hi*8 + p, n = lane&15)
  #pragma unroll
  for (int hm = 0; hm < 2; ++hm)
    #pragma unroll
    for (int nt = 0; nt < 4; ++nt)
      #pragma unroll
      for (int p = 0; p < 8; ++p)
        C[(size_t)(m0 + w * 32 + hm * 16 + hi * 8 + p) * N + n0 + nt * 16 + r] =
            acc[hm][nt][p];
}

// ---------------------------------------------------------------------------
// RoPE + layout transform: fp32 token-major [B,T,H*64] -> bf16 head-major
// [B,H,T,64].  grid = (T, B*H), block = 32 (one pair per thread).
// ---------------------------------------------------------------------------
__global__ __launch_bounds__(32)
void mha_rope_cvt(const float* __restrict__ src, __bf16* __restrict__ dst,
                  const int* __restrict__ pos, int rotate) {
  const int t  = blockIdx.x;
  const int bh = blockIdx.y;          // b*H + h
  const int b  = bh >> 4, h = bh & 15;
  const int i  = threadIdx.x;         // pair index 0..31

  const float* s = src + ((size_t)(b * TT + t)) * DD + h * DH;
  float x1 = s[2 * i], x2 = s[2 * i + 1];
  float r1 = x1, r2 = x2;
  if (rotate) {
    // inv_freq = theta^(-i/32), theta = 10000  (ln 10000 = 9.210340372)
    float inv = __expf(-(float)i * (9.210340372f / 32.0f));
    float ang = (float)pos[t] * inv;
    float sn = __sinf(ang), cs = __cosf(ang);
    r1 = x1 * cs - x2 * sn;
    r2 = x1 * sn + x2 * cs;
  }
  __bf16* d = dst + ((size_t)bh * TT + t) * DH + 2 * i;
  d[0] = (__bf16)r1;
  d[1] = (__bf16)r2;
}

// ---------------------------------------------------------------------------
// Flash attention: one wave per (b,h, 16-query tile); stream 32-key chunks.
// V tile staged via Tensor Data Mover (async, TENSORcnt) when available,
// overlapped with the QK^T WMMAs.  S via 4 WMMA, online softmax,
// O += P V via 4 WMMA per chunk.  Output bf16 token-major [B,T,H*64].
// ---------------------------------------------------------------------------
__global__ __launch_bounds__(32)
void mha_flash_attn(const __bf16* __restrict__ Qh, const __bf16* __restrict__ Kh,
                    const __bf16* __restrict__ Vh, __bf16* __restrict__ AO) {
  __shared__ __align__(16) __bf16 sV[32 * 64];  // V chunk, row-major [key][dh]
  __shared__ __align__(16) __bf16 sP[16 * 32];  // probabilities [m][key-in-chunk]

  const int lane = threadIdx.x;
  const int r  = lane & 15;
  const int hi = lane >> 4;
  const int q0 = blockIdx.x * 16;
  const int bh = blockIdx.y;

  const __bf16* Qb = Qh + ((size_t)bh * TT + q0) * DH;
  const __bf16* Kb = Kh + (size_t)bh * TT * DH;
  const __bf16* Vb = Vh + (size_t)bh * TT * DH;

  // Q tile in registers: 16x64 = two 16x32 A-fragments
  v16bf qf0 = frag_row(Qb + r * DH, hi);
  v16bf qf1 = frag_row(Qb + r * DH + 32, hi);

  v8f o[4] = {};
  float mrow[8], lrow[8];
  #pragma unroll
  for (int p = 0; p < 8; ++p) { mrow[p] = -1e30f; lrow[p] = 0.0f; }

  const int nch = (q0 + 47) >> 5;   // chunks of 32 keys covering [0, q0+16)
  for (int ch = 0; ch < nch; ++ch) {
    const int kc = ch * 32;

    // ---- stage V chunk (32x64 bf16 = 4 KiB) into LDS -------------------
#if MHA_USE_TDM
    // async DMA via TDM; previous-iteration sV reads were fenced by the
    // trailing __syncthreads (s_wait_dscnt 0).
    tdm_load_rows64(Vb + (size_t)kc * DH, (unsigned)(__SIZE_TYPE__)sV, 32);
#else
    #pragma unroll
    for (int it = 0; it < 8; ++it) {
      int c = it * 32 + lane;       // 0..255 chunks of 8 elems
      *(v8bf*)(sV + c * 8) = *(const v8bf*)(Vb + (size_t)kc * DH + c * 8);
    }
#endif

    // ---- S = Q K^T : two 16x16 tiles (overlaps with the V DMA) ---------
    v8f s0 = {}, s1 = {};
    {
      const __bf16* kr0 = Kb + (size_t)(kc + r) * DH;
      s0 = wmma_bf16(qf0, frag_row(kr0, hi), s0);
      s0 = wmma_bf16(qf1, frag_row(kr0 + 32, hi), s0);
      const __bf16* kr1 = Kb + (size_t)(kc + 16 + r) * DH;
      s1 = wmma_bf16(qf0, frag_row(kr1, hi), s1);
      s1 = wmma_bf16(qf1, frag_row(kr1 + 32, hi), s1);
    }

    // ---- online softmax (row m = hi*8 + p, replicated over 16 lanes) ---
    #pragma unroll
    for (int p = 0; p < 8; ++p) {
      int qg = q0 + hi * 8 + p;
      float v0 = (kc + r      <= qg) ? s0[p] * 0.125f : -1e30f; // 1/sqrt(64)
      float v1 = (kc + 16 + r <= qg) ? s1[p] * 0.125f : -1e30f;
      float rm = rmax16(fmaxf(v0, v1));
      float mn = fmaxf(mrow[p], rm);
      float alpha = __expf(mrow[p] - mn);
      float e0 = __expf(v0 - mn);
      float e1 = __expf(v1 - mn);
      lrow[p] = lrow[p] * alpha + rsum16(e0 + e1);
      mrow[p] = mn;
      #pragma unroll
      for (int nt = 0; nt < 4; ++nt) o[nt][p] *= alpha;
      int m = hi * 8 + p;
      sP[m * 32 + r]      = (__bf16)e0;
      sP[m * 32 + 16 + r] = (__bf16)e1;
    }
    __syncthreads();                      // sP visible
#if MHA_USE_TDM
    __builtin_amdgcn_s_wait_tensorcnt((short)0);   // V DMA complete
#endif

    // ---- O += P(16x32) x V(32x64) --------------------------------------
    v16bf pf = frag_row(sP + r * 32, hi);
    #pragma unroll
    for (int nt = 0; nt < 4; ++nt) {
      int cidx = nt * 16 + r;            // B-fragment: lane = column of V
      v16bf vf;
      #pragma unroll
      for (int j = 0; j < 8; ++j) {
        vf[j]     = sV[(hi * 8 + j) * 64 + cidx];
        vf[8 + j] = sV[(16 + hi * 8 + j) * 64 + cidx];
      }
      o[nt] = wmma_bf16(pf, vf, o[nt]);
    }
    __syncthreads();   // sV/sP reads done before next chunk restages
  }

  // normalize + write bf16 token-major [B,T,1024]
  const int b = bh >> 4, h = bh & 15;
  #pragma unroll
  for (int p = 0; p < 8; ++p) {
    int m = hi * 8 + p;
    float inv = 1.0f / lrow[p];
    size_t rowoff = ((size_t)(b * TT + q0 + m)) * DD + h * DH;
    #pragma unroll
    for (int nt = 0; nt < 4; ++nt)
      AO[rowoff + nt * 16 + r] = (__bf16)(o[nt][p] * inv);
  }
}

// ---------------------------------------------------------------------------
extern "C" void kernel_launch(void* const* d_in, const int* in_sizes, int n_in,
                              void* d_out, int out_size, void* d_ws,
                              size_t ws_size, hipStream_t stream) {
  const float* X  = (const float*)d_in[0];
  const float* Wq = (const float*)d_in[1];
  const float* Wk = (const float*)d_in[2];
  const float* Wv = (const float*)d_in[3];
  const float* Wo = (const float*)d_in[4];
  const int*  pos = (const int*)d_in[5];
  float* out = (float*)d_out;

  char* ws = (char*)d_ws;
  size_t off = 0;
  auto alloc = [&](size_t bytes) -> void* {
    void* p = ws + off;
    off += (bytes + 255) & ~(size_t)255;
    return p;
  };

  __bf16* Xb  = (__bf16*)alloc((size_t)MM * DD * 2);   // 8 MiB
  __bf16* Wqb = (__bf16*)alloc((size_t)DD * DD * 2);
  __bf16* Wkb = (__bf16*)alloc((size_t)DD * DD * 2);
  __bf16* Wvb = (__bf16*)alloc((size_t)DD * DD * 2);
  __bf16* Wob = (__bf16*)alloc((size_t)DD * DD * 2);
  float*  Qf  = (float*) alloc((size_t)MM * DD * 4);   // 16 MiB each
  float*  Kf  = (float*) alloc((size_t)MM * DD * 4);
  float*  Vf  = (float*) alloc((size_t)MM * DD * 4);
  __bf16* Qh  = (__bf16*)alloc((size_t)MM * DD * 2);   // head-major
  __bf16* Kh  = (__bf16*)alloc((size_t)MM * DD * 2);
  __bf16* Vh  = (__bf16*)alloc((size_t)MM * DD * 2);
  __bf16* AOb = (__bf16*)alloc((size_t)MM * DD * 2);

  // 1) fp32 -> bf16
  {
    int nx = MM * DD, nw = DD * DD;
    mha_cvt_bf16<<<(nx + 255) / 256, 256, 0, stream>>>(X, Xb, nx);
    mha_cvt_bf16<<<(nw + 255) / 256, 256, 0, stream>>>(Wq, Wqb, nw);
    mha_cvt_bf16<<<(nw + 255) / 256, 256, 0, stream>>>(Wk, Wkb, nw);
    mha_cvt_bf16<<<(nw + 255) / 256, 256, 0, stream>>>(Wv, Wvb, nw);
    mha_cvt_bf16<<<(nw + 255) / 256, 256, 0, stream>>>(Wo, Wob, nw);
  }

  // 2) Q/K/V projections (X @ W^T), WMMA
  dim3 ggrid(DD / 64, MM / 128);
  mha_gemm_bf16<<<ggrid, 128, 0, stream>>>(Xb, Wqb, Qf, MM, DD, DD);
  mha_gemm_bf16<<<ggrid, 128, 0, stream>>>(Xb, Wkb, Kf, MM, DD, DD);
  mha_gemm_bf16<<<ggrid, 128, 0, stream>>>(Xb, Wvb, Vf, MM, DD, DD);

  // 3) RoPE (Q,K) + head-major bf16 repack (Q,K,V)
  dim3 rgrid(TT, BB * HH);
  mha_rope_cvt<<<rgrid, 32, 0, stream>>>(Qf, Qh, pos, 1);
  mha_rope_cvt<<<rgrid, 32, 0, stream>>>(Kf, Kh, pos, 1);
  mha_rope_cvt<<<rgrid, 32, 0, stream>>>(Vf, Vh, pos, 0);

  // 4) causal flash attention (WMMA + TDM async V staging)
  dim3 agrid(TT / 16, BB * HH);
  mha_flash_attn<<<agrid, 32, 0, stream>>>(Qh, Kh, Vh, AOb);

  // 5) output projection -> fp32 d_out
  mha_gemm_bf16<<<ggrid, 128, 0, stream>>>(AOb, Wob, out, MM, DD, DD);
}
